// regression_model_41996190220463
// MI455X (gfx1250) — matrix operand order repaired and checked by reading, not compiled, not run on hardware
//
#include <hip/hip_runtime.h>
#include <hip/hip_bf16.h>
#include <stdint.h>

typedef _Float16 half_t;
typedef _Float16 h8   __attribute__((ext_vector_type(8)));
typedef _Float16 v16h __attribute__((ext_vector_type(16)));
typedef float    v8f  __attribute__((ext_vector_type(8)));
typedef uint32_t u32x4 __attribute__((ext_vector_type(4)));
typedef int      i32x4 __attribute__((ext_vector_type(4)));
typedef int      i32x8 __attribute__((ext_vector_type(8)));

#if __has_include(<hip/amd_detail/amd_gfx1250_TDM.h>)
#define TDM_6ARG 1
#endif

#define BATCH 8
constexpr int ROWS = 16128;   // stage activation buffer rows (16000 + 128 pad)
constexpr int T0   = 16000;   // sinc output length
constexpr int NWAV = 48250;

// ---------------------------------------------------------------- helpers
__device__ __forceinline__ float sigmf(float x){ return 1.f/(1.f+expf(-x)); }

__global__ void k_zero(uint32_t* p, size_t n){
  size_t i = (size_t)blockIdx.x*blockDim.x + threadIdx.x;
  if (i < n) p[i] = 0u;
}

__global__ void k_zero_rows(half_t* base, size_t batchStride, size_t rowStart, size_t cnt){
  size_t i = (size_t)blockIdx.x*blockDim.x + threadIdx.x;
  size_t b = i / cnt, k = i % cnt;
  if (b >= BATCH) return;
  base[b*batchStride + rowStart*64 + k] = (half_t)0.f;
}

__global__ void k_cvt(const float* __restrict__ s, half_t* __restrict__ d, size_t n){
  size_t i = (size_t)blockIdx.x*blockDim.x + threadIdx.x;
  if (i < n) d[i] = (half_t)s[i];
}

// OIH f32 -> [KW][O][I] f16
__global__ void k_wconv(const float* __restrict__ src, half_t* __restrict__ dst,
                        int O, int I, int KW){
  int i = blockIdx.x*blockDim.x + threadIdx.x;
  int total = O*I*KW;
  if (i >= total) return;
  int kw = i % KW; int ii = (i/KW) % I; int o = i/(KW*I);
  dst[((size_t)kw*O + o)*I + ii] = (half_t)src[((size_t)o*I + ii)*KW + kw];
}

__global__ void k_bias_sum(const float* a, const float* b, float* o, int n){
  int i = blockIdx.x*blockDim.x + threadIdx.x;
  if (i < n) o[i] = a[i] + b[i];
}

__global__ void k_lens(const int* lenX, const int* lenLM, int* f, int* l){
  int b = threadIdx.x;
  if (b >= BATCH) return;
  int v = (lenX[b]-250)/3;
  v/=4; v/=5; v/=2; v/=2;
  f[b]=v; l[b]=v+lenLM[b];
}

// ---------------------------------------------------------------- WMMA GEMM
// D[N][M] (+bias, act) = sum_j A_j[M][K] * B[n + j*dil][K]^T ; ldb == K.
// Each wave computes a 32(M)x16(N) tile (2 accumulators share one B fragment).
// When useLds != 0 the whole [taps][M][K] weight tile (<= 64KB) is DMA-staged
// into LDS by wave 0 via the Tensor Data Mover, then read with ds loads.
__global__ __launch_bounds__(256)
void k_gemm(const half_t* __restrict__ A, const half_t* __restrict__ B,
            void* __restrict__ Dv, const float* __restrict__ bias,
            int M, int K, int N, int taps, int dil,
            long long sA, long long sB, long long sD,
            int act, int outF32, int accum, int useLds)
{
  __shared__ half_t As[32768];          // 64 KB weight stage
  int b = blockIdx.z;
  A += (size_t)(sA * b);
  B += (size_t)(sB * b);
  int wave = threadIdx.x >> 5;
  int lane = threadIdx.x & 31;

  if (useLds){
    if (wave == 0){
      uint64_t ga = (uint64_t)(uintptr_t)A;
      uint32_t lds = (uint32_t)(uintptr_t)&As[0];
      uint32_t E = (uint32_t)(((size_t)taps*M*K) >> 2);   // in 8-byte units
      // D# group0: count=1, lds_addr, global_addr[56:0], type=2
      u32x4 g0;
      g0[0] = 1u;
      g0[1] = lds;
      g0[2] = (uint32_t)ga;
      g0[3] = (uint32_t)((ga>>32) & 0x01FFFFFFu) | (2u<<30);
      // D# group1: data_size=8B, 1-D tensor of E elements, tile = whole tensor
      i32x8 g1;
      g1[0] = (int)(3u<<16);                       // data_size = 3 (8B)
      g1[1] = (int)((E & 0xFFFFu) << 16);          // tensor_dim0 [15:0]
      g1[2] = (int)(((E>>16) & 0xFFFFu) | (1u<<16)); // tensor_dim0 hi | tensor_dim1=1
      g1[3] = (int)((E & 0xFFFFu) << 16);          // tile_dim0 = E
      g1[4] = 1;                                    // tile_dim1 = 1
      g1[5] = (int)E;                               // tensor_dim0_stride lo
      g1[6] = 0;
      g1[7] = 0;
      i32x4 z4; z4[0]=0; z4[1]=0; z4[2]=0; z4[3]=0;
#ifdef TDM_6ARG
      i32x8 z8; for (int i=0;i<8;++i) z8[i]=0;
      __builtin_amdgcn_tensor_load_to_lds(g0, g1, z4, z4, z8, 0);
#else
      __builtin_amdgcn_tensor_load_to_lds(g0, g1, z4, z4, 0);
#endif
      __builtin_amdgcn_s_wait_tensorcnt(0);
    }
    __syncthreads();
  }

  int tM0 = (blockIdx.y*2 + (wave>>2))*2;
  int tM1 = tM0 + 1;
  int tN  = blockIdx.x*4 + (wave&3);
  bool v1 = (tM1*16 < M);
  if (tM0*16 >= M || tN*16 >= N) return;
  int mrow0 = tM0*16 + (lane&15);
  int mrow1 = v1 ? (tM1*16 + (lane&15)) : mrow0;
  int ncol  = tN*16 + (lane&15);
  int hi    = (lane>>4)&1;
  v8f acc0 = {0.f,0.f,0.f,0.f,0.f,0.f,0.f,0.f};
  v8f acc1 = {0.f,0.f,0.f,0.f,0.f,0.f,0.f,0.f};

#define GEMM_LOOP(APTR)                                                       \
  for (int j=0;j<taps;++j){                                                   \
    const half_t* Ar0 = APTR + (size_t)j*M*K + (size_t)mrow0*K;               \
    const half_t* Ar1 = APTR + (size_t)j*M*K + (size_t)mrow1*K;               \
    const half_t* Br  = B + (size_t)j*dil*K + (size_t)ncol*K;                 \
    for (int kt=0;kt<K;kt+=32){                                               \
      int kA = kt + hi*8;                                                     \
      int kB = kt + hi*8;                                                     \
      h8 a00 = *(const h8*)(Ar0 + kA);                                        \
      h8 a01 = *(const h8*)(Ar0 + kA + 16);                                   \
      h8 a10 = *(const h8*)(Ar1 + kA);                                        \
      h8 a11 = *(const h8*)(Ar1 + kA + 16);                                   \
      h8 b0  = *(const h8*)(Br + kB);                                         \
      h8 b1  = *(const h8*)(Br + kB + 16);                                    \
      if (kt + 32 < K) __builtin_prefetch((const void*)(Br + kt + 32), 0, 1); \
      v16h av0, av1, bv;                                                      \
      _Pragma("unroll")                                                       \
      for (int i=0;i<8;++i){ av0[i]=a00[i]; av0[i+8]=a01[i];                  \
                             av1[i]=a10[i]; av1[i+8]=a11[i];                  \
                             bv[i]=b0[i];   bv[i+8]=b1[i]; }                  \
      acc0 = __builtin_amdgcn_wmma_f32_16x16x32_f16(false, av0, false, bv,    \
                                                    (short)0, acc0, false, false); \
      acc1 = __builtin_amdgcn_wmma_f32_16x16x32_f16(false, av1, false, bv,    \
                                                    (short)0, acc1, false, false); \
    }                                                                         \
  }

  if (useLds){ GEMM_LOOP(As) } else { GEMM_LOOP(A) }
#undef GEMM_LOOP

  if (ncol >= N) return;
  auto store_tile = [&](const v8f& acc, int tM){
    int mb = tM*16 + hi*8;
    float vals[8];
    #pragma unroll
    for (int v=0;v<8;++v){
      float x = acc[v];
      if (bias) x += bias[mb+v];
      if (act==1) x = x>0.f ? x : 0.f;
      vals[v]=x;
    }
    if (outF32){
      float* Dp = (float*)Dv + (size_t)(sD*b) + (size_t)ncol*M + mb;
      if (accum){ for (int v=0;v<8;++v) Dp[v]+=vals[v]; }
      else      { for (int v=0;v<8;++v) Dp[v]=vals[v]; }
    } else {
      half_t* Dp = (half_t*)Dv + (size_t)(sD*b) + (size_t)ncol*M + mb;
      if (accum){
        for (int v=0;v<8;++v) Dp[v] = (half_t)((float)Dp[v] + vals[v]);
      } else {
        h8 o;
        #pragma unroll
        for (int v=0;v<8;++v) o[v] = (half_t)vals[v];
        *(h8*)Dp = o;
      }
    }
  };
  store_tile(acc0, tM0);
  if (v1) store_tile(acc1, tM1);
}

// ---------------------------------------------------------------- front end
// sinc conv: x[B][48250] f32 -> y[B][ROWS][64] f16, stride 3, k=251, leaky 0.2
__global__ void k_sinc(const float* __restrict__ x, const float* __restrict__ w,
                       const float* __restrict__ bias, half_t* __restrict__ y)
{
  size_t idx = (size_t)blockIdx.x*blockDim.x + threadIdx.x;
  if (idx >= (size_t)BATCH*T0*64) return;
  int c = idx & 63;
  int t = (int)((idx>>6) % T0);
  int b = (int)(idx / ((size_t)64*T0));
  const float* xs = x + (size_t)b*NWAV + t*3;
  const float* wc = w + (size_t)c*251;
  float acc = bias[c];
  for (int k=0;k<251;++k) acc = fmaf(wc[k], xs[k], acc);
  acc = acc>0.f ? acc : 0.2f*acc;
  y[((size_t)b*ROWS + t)*64 + c] = (half_t)acc;
}

// gated activation: g[B][ROWS][128] -> u[B][ROWS][64]
__global__ void k_gate(const half_t* __restrict__ g, half_t* __restrict__ u, int T){
  size_t idx = (size_t)blockIdx.x*blockDim.x + threadIdx.x;
  size_t per = (size_t)T*64;
  if (idx >= (size_t)BATCH*per) return;
  size_t b = idx / per, r = idx % per;
  int t = (int)(r>>6), c = (int)(r&63);
  float a = (float)g[(b*ROWS + t)*128 + c];
  float s = (float)g[(b*ROWS + t)*128 + c + 64];
  u[(b*ROWS + t)*64 + c] = (half_t)(tanhf(a) * sigmf(s));
}

__global__ void k_resid(half_t* __restrict__ x, const half_t* __restrict__ r, int T){
  size_t idx = (size_t)blockIdx.x*blockDim.x + threadIdx.x;
  size_t per = (size_t)T*64;
  if (idx >= (size_t)BATCH*per) return;
  size_t b = idx / per, q = idx % per;
  size_t o = b*(size_t)ROWS*64 + q;
  x[o] = (half_t)(((float)r[o] + (float)x[o]) * 0.707f);
}

__global__ void k_relu16(half_t* __restrict__ x, int T){
  size_t idx = (size_t)blockIdx.x*blockDim.x + threadIdx.x;
  size_t per = (size_t)T*64;
  if (idx >= (size_t)BATCH*per) return;
  size_t b = idx / per, q = idx % per;
  size_t o = b*(size_t)ROWS*64 + q;
  float v = (float)x[o];
  x[o] = (half_t)(v>0.f?v:0.f);
}

__global__ void k_maxpool(const half_t* __restrict__ in, half_t* __restrict__ out,
                          int Tout, int k){
  size_t idx = (size_t)blockIdx.x*blockDim.x + threadIdx.x;
  size_t per = (size_t)Tout*64;
  if (idx >= (size_t)BATCH*per) return;
  size_t b = idx / per, q = idx % per;
  int t = (int)(q>>6), c = (int)(q&63);
  float m = -3.0e38f;
  for (int i=0;i<k;++i){
    float v = (float)in[(b*ROWS + (size_t)t*k + i)*64 + c];
    m = fmaxf(m, v);
  }
  out[(b*ROWS + t)*64 + c] = (half_t)m;
}

// ---------------------------------------------------------------- LSTM
// xW[B][208][512] f32 (i,f,g,o), Whh f16 [512][128] -> hs[B][200][128]
__global__ __launch_bounds__(128)
void k_lstm(const float* __restrict__ xW, const half_t* __restrict__ Whh,
            float* __restrict__ hs, int T)
{
  int b = blockIdx.x, j = threadIdx.x;
  __shared__ float hsm[128];
  float c = 0.f;
  hsm[j] = 0.f;
  __syncthreads();
  const half_t* wi = Whh + (size_t)j*128;
  const half_t* wf = Whh + (size_t)(j+128)*128;
  const half_t* wg = Whh + (size_t)(j+256)*128;
  const half_t* wo = Whh + (size_t)(j+384)*128;
  for (int t=0;t<T;++t){
    const float* xw = xW + ((size_t)b*208 + t)*512;
    float gi = xw[j], gf = xw[j+128], gg = xw[j+256], go = xw[j+384];
    for (int k=0;k<128;++k){
      float h = hsm[k];
      gi = fmaf((float)wi[k], h, gi);
      gf = fmaf((float)wf[k], h, gf);
      gg = fmaf((float)wg[k], h, gg);
      go = fmaf((float)wo[k], h, go);
    }
    c = sigmf(gf)*c + sigmf(gi)*tanhf(gg);
    float h = sigmf(go)*tanhf(c);
    __syncthreads();
    hsm[j] = h;
    __syncthreads();
    hs[((size_t)b*T + t)*128 + j] = h;
  }
}

// reversal on the valid prefix: yr[t] = y[clip(f-1-t,0,199)]
__global__ void k_rev(const half_t* __restrict__ y, half_t* __restrict__ yr,
                      const int* __restrict__ f){
  size_t idx = (size_t)blockIdx.x*blockDim.x + threadIdx.x;
  if (idx >= (size_t)BATCH*200*64) return;
  int c = (int)(idx & 63);
  int t = (int)((idx>>6) % 200);
  int b = (int)(idx / (200*64));
  int rv = f[b]-1-t; rv = rv<0?0:(rv>199?199:rv);
  yr[((size_t)b*208 + t)*64 + c] = y[((size_t)b*ROWS + rv)*64 + c];
}

__global__ void k_unrev(const float* __restrict__ hbr, float* __restrict__ hb,
                        const int* __restrict__ f){
  size_t idx = (size_t)blockIdx.x*blockDim.x + threadIdx.x;
  if (idx >= (size_t)BATCH*200*128) return;
  int d = (int)(idx & 127);
  int t = (int)((idx>>7) % 200);
  int b = (int)(idx / (200*128));
  int rv = f[b]-1-t; rv = rv<0?0:(rv>199?199:rv);
  hb[((size_t)b*200 + t)*128 + d] = hbr[((size_t)b*200 + rv)*128 + d];
}

// ---------------------------------------------------------------- fusion head
// n[B][368][256] f32 + f16 copy: rows <f: [hf|hb]; rows <f+lw: wemb; else 0
__global__ void k_build_n(const float* __restrict__ hf, const float* __restrict__ hb,
                          const float* __restrict__ wemb, const int* __restrict__ f,
                          const int* __restrict__ lm, float* __restrict__ n,
                          half_t* __restrict__ nh){
  size_t idx = (size_t)blockIdx.x*blockDim.x + threadIdx.x;
  if (idx >= (size_t)BATCH*368*256) return;
  int d = (int)(idx & 255);
  int r = (int)((idx>>8) % 368);
  int b = (int)(idx / (368*256));
  int F = f[b], LW = lm[b];
  float v = 0.f;
  if (r < F){
    v = (d<128) ? hf[((size_t)b*200 + r)*128 + d]
                : hb[((size_t)b*200 + r)*128 + (d-128)];
  } else if (r < F + LW){
    v = wemb[((size_t)b*160 + (r-F))*256 + d];
  }
  size_t o = ((size_t)b*368 + r)*256 + d;
  n[o] = v;
  nh[o] = (half_t)v;
}

// masked softmax over axis 2 (row l), accumulate column sums into c1
__global__ __launch_bounds__(256)
void k_rowsm(const float* __restrict__ sc, const int* __restrict__ l1,
             const int* __restrict__ l2, float* __restrict__ c1){
  int b = blockIdx.y, l = blockIdx.x;
  int L1 = l1[b], L2 = l2[b];
  if (l >= L1) return;
  const float* row = sc + ((size_t)b*368 + l)*368;
  __shared__ float red[256];
  float mx = -3.0e38f;
  for (int m=threadIdx.x; m<368; m+=256) if (m<L2) mx = fmaxf(mx,row[m]);
  red[threadIdx.x]=mx; __syncthreads();
  for (int s=128;s>0;s>>=1){ if((int)threadIdx.x<s) red[threadIdx.x]=fmaxf(red[threadIdx.x],red[threadIdx.x+s]); __syncthreads(); }
  mx = red[0]; __syncthreads();
  float sum = 0.f;
  for (int m=threadIdx.x; m<368; m+=256) if (m<L2) sum += expf(row[m]-mx);
  red[threadIdx.x]=sum; __syncthreads();
  for (int s=128;s>0;s>>=1){ if((int)threadIdx.x<s) red[threadIdx.x]+=red[threadIdx.x+s]; __syncthreads(); }
  float inv = 1.f/red[0];
  for (int m=threadIdx.x; m<368; m+=256)
    if (m<L2) atomicAdd(&c1[b*368+m], expf(row[m]-mx)*inv);
}

// masked softmax over axis 1 (column m), accumulate row sums into c2
__global__ __launch_bounds__(256)
void k_colsm(const float* __restrict__ sc, const int* __restrict__ l1,
             const int* __restrict__ l2, float* __restrict__ c2){
  int b = blockIdx.y, m = blockIdx.x;
  int L1 = l1[b], L2 = l2[b];
  if (m >= L2) return;
  const float* base = sc + (size_t)b*368*368 + m;
  __shared__ float red[256];
  float mx = -3.0e38f;
  for (int l=threadIdx.x; l<368; l+=256) if (l<L1) mx = fmaxf(mx, base[(size_t)l*368]);
  red[threadIdx.x]=mx; __syncthreads();
  for (int s=128;s>0;s>>=1){ if((int)threadIdx.x<s) red[threadIdx.x]=fmaxf(red[threadIdx.x],red[threadIdx.x+s]); __syncthreads(); }
  mx = red[0]; __syncthreads();
  float sum = 0.f;
  for (int l=threadIdx.x; l<368; l+=256) if (l<L1) sum += expf(base[(size_t)l*368]-mx);
  red[threadIdx.x]=sum; __syncthreads();
  for (int s=128;s>0;s>>=1){ if((int)threadIdx.x<s) red[threadIdx.x]+=red[threadIdx.x+s]; __syncthreads(); }
  float inv = 1.f/red[0];
  for (int l=threadIdx.x; l<368; l+=256)
    if (l<L1) atomicAdd(&c2[b*368+l], expf(base[(size_t)l*368]-mx)*inv);
}

// dout[d] = | sum_r coef[r]*wmat[r][d] - sum_r smat[r][d] | / len
__global__ __launch_bounds__(256)
void k_dvec(const float* __restrict__ coef, const float* __restrict__ wmat,
            const float* __restrict__ smat, const int* __restrict__ len,
            float* __restrict__ dout){
  int b = blockIdx.x, d = threadIdx.x;
  float acc=0.f, accs=0.f;
  for (int r=0;r<368;++r){
    size_t o = ((size_t)b*368 + r)*256 + d;
    acc  = fmaf(coef[b*368+r], wmat[o], acc);
    accs += smat[o];
  }
  dout[b*256+d] = fabsf(acc - accs) / (float)len[b];
}

__global__ __launch_bounds__(128)
void k_head(const float* __restrict__ dv, const float* __restrict__ W1,
            const float* __restrict__ b1, const float* __restrict__ W2,
            const float* __restrict__ b2, float* __restrict__ o){
  int b = blockIdx.x, j = threadIdx.x;
  __shared__ float hs[128];
  float acc = b1[j];
  const float* w = W1 + (size_t)j*256;
  const float* x = dv + (size_t)b*256;
  for (int k=0;k<256;++k) acc = fmaf(w[k], x[k], acc);
  hs[j] = fmaxf(acc,0.f) * W2[j];
  __syncthreads();
  for (int s=64;s>0;s>>=1){ if (j<s) hs[j]+=hs[j+s]; __syncthreads(); }
  if (j==0) o[b] = hs[0] + b2[0];
}

__global__ void k_final(const float* o1, const float* o2, float* out){
  int b = threadIdx.x;
  if (b < BATCH) out[b] = 0.5f*(o1[b]+o2[b]);
}

// ================================================================ host
extern "C" void kernel_launch(void* const* d_in, const int* in_sizes, int n_in,
                              void* d_out, int out_size, void* d_ws, size_t ws_size,
                              hipStream_t stream)
{
  (void)in_sizes; (void)n_in; (void)out_size; (void)ws_size;
  const float* x1 = (const float*)d_in[0];
  const float* x2 = (const float*)d_in[1];
  const float* wavlm1 = (const float*)d_in[2];
  const float* wavlm2 = (const float*)d_in[3];

  // params flattened in setup_inputs insertion order after the 4 arrays
  const int PB = 4;
  const int SINC_W = PB+0, SINC_B = PB+1;
  auto WN = [&](int s, int off){ return PB + 2 + s*48 + off; };  // per-stage block
  const int LSTM0 = PB + 2 + 4*48;       // fw: Wih,Whh,bih,bhh ; bw: +4
  const int LM0   = LSTM0 + 8;           // W1,b1,W2,b2
  const int DV0   = LM0 + 4;             // W1,b1,W2,b2
  const int LENX1 = DV0 + 4, LENX2 = LENX1+1, LENLM1 = LENX1+2, LENLM2 = LENX1+3;
  auto PF = [&](int i)->const float* { return (const float*)d_in[i]; };
  auto PI = [&](int i)->const int*   { return (const int*)d_in[i]; };

  uint8_t* ws = (uint8_t*)d_ws;
  size_t off = 0;
  auto alloc = [&](size_t bytes)->size_t{
    off = (off + 255) & ~(size_t)255;
    size_t r = off; off += bytes; return r;
  };

  // ---- f16 weight arena
  size_t WOFF = alloc((size_t)4<<20);
  half_t* WH = (half_t*)(ws + WOFF);
  size_t woff = 0;
  auto cvtw = [&](int pidx, int O, int I, int KW)->size_t{
    size_t o = woff; woff += ((size_t)O*I*KW + 127) & ~(size_t)127;
    int total = O*I*KW;
    k_wconv<<<(total+255)/256,256,0,stream>>>(PF(pidx), WH+o, O, I, KW);
    return o;
  };
  size_t st_first[4], st_in[4][7], st_skip[4][7], st_res[4][7], st_lc1[4], st_lc2[4];
  for (int s=0;s<4;++s){
    st_first[s] = cvtw(WN(s,0), 64,64,1);
    for (int k=0;k<7;++k){
      st_in[s][k]   = cvtw(WN(s,2+6*k+0), 128,64,3);
      st_skip[s][k] = cvtw(WN(s,2+6*k+2), 64,64,1);
      st_res[s][k]  = cvtw(WN(s,2+6*k+4), 64,64,1);
    }
    st_lc1[s] = cvtw(WN(s,44), 64,64,1);
    st_lc2[s] = cvtw(WN(s,46), 64,64,1);
  }
  size_t w_ih_f = cvtw(LSTM0+0, 512,64,1);
  size_t w_hh_f = cvtw(LSTM0+1, 512,128,1);
  size_t w_ih_b = cvtw(LSTM0+4, 512,64,1);
  size_t w_hh_b = cvtw(LSTM0+5, 512,128,1);
  size_t w_lm1  = cvtw(LM0+0, 256,768,1);
  size_t w_lm2  = cvtw(LM0+2, 256,256,1);

  // ---- small state
  size_t BSF = alloc(512*4), BSB = alloc(512*4);
  k_bias_sum<<<2,256,0,stream>>>(PF(LSTM0+2), PF(LSTM0+3), (float*)(ws+BSF), 512);
  k_bias_sum<<<2,256,0,stream>>>(PF(LSTM0+6), PF(LSTM0+7), (float*)(ws+BSB), 512);

  size_t F1 = alloc(32), F2 = alloc(32), L1 = alloc(32), L2 = alloc(32);
  k_lens<<<1,32,0,stream>>>(PI(LENX1), PI(LENLM1), (int*)(ws+F1), (int*)(ws+L1));
  k_lens<<<1,32,0,stream>>>(PI(LENX2), PI(LENLM2), (int*)(ws+F2), (int*)(ws+L2));

  // ---- activation / intermediate buffers
  const size_t RB64 = (size_t)BATCH*ROWS*64*2;
  size_t SINo = alloc(RB64), XBo = alloc(RB64), Uo = alloc(RB64),
         SKo = alloc(RB64), RTo = alloc(RB64);
  size_t Go   = alloc((size_t)BATCH*ROWS*128*2);
  half_t *SIN=(half_t*)(ws+SINo), *XB=(half_t*)(ws+XBo), *U=(half_t*)(ws+Uo),
         *SK=(half_t*)(ws+SKo), *RT=(half_t*)(ws+RTo), *G=(half_t*)(ws+Go);
  size_t Y4Ro = alloc((size_t)BATCH*208*64*2);
  size_t XWo  = alloc((size_t)BATCH*208*512*4);
  size_t HBRo = alloc((size_t)BATCH*200*128*4);
  size_t HF1o = alloc((size_t)BATCH*200*128*4), HB1o = alloc((size_t)BATCH*200*128*4);
  size_t HF2o = alloc((size_t)BATCH*200*128*4), HB2o = alloc((size_t)BATCH*200*128*4);
  size_t WLHo = alloc((size_t)BATCH*160*768*2);
  size_t LMHo = alloc((size_t)BATCH*160*256*2);
  size_t WEMB1o = alloc((size_t)BATCH*160*256*4), WEMB2o = alloc((size_t)BATCH*160*256*4);
  size_t N1o = alloc((size_t)BATCH*368*256*4), N2o = alloc((size_t)BATCH*368*256*4);
  size_t N1Ho = alloc((size_t)BATCH*368*256*2), N2Ho = alloc((size_t)BATCH*368*256*2);
  size_t SCo = alloc((size_t)BATCH*368*368*4);
  size_t C1o = alloc((size_t)BATCH*368*4), C2o = alloc((size_t)BATCH*368*4);
  size_t D1o = alloc((size_t)BATCH*256*4), D2o = alloc((size_t)BATCH*256*4);
  size_t O1o = alloc(64), O2o = alloc(64);

  half_t* Y4R=(half_t*)(ws+Y4Ro); float* XW=(float*)(ws+XWo);
  float* HBR=(float*)(ws+HBRo);
  float *HF1=(float*)(ws+HF1o), *HB1=(float*)(ws+HB1o),
        *HF2=(float*)(ws+HF2o), *HB2=(float*)(ws+HB2o);
  half_t* WLH=(half_t*)(ws+WLHo); half_t* LMH=(half_t*)(ws+LMHo);
  float *WEMB1=(float*)(ws+WEMB1o), *WEMB2=(float*)(ws+WEMB2o);
  float *N1=(float*)(ws+N1o), *N2=(float*)(ws+N2o);
  half_t *N1H=(half_t*)(ws+N1Ho), *N2H=(half_t*)(ws+N2Ho);
  float* SC=(float*)(ws+SCo);
  float *C1=(float*)(ws+C1o), *C2=(float*)(ws+C2o);
  float *D1=(float*)(ws+D1o), *D2=(float*)(ws+D2o);
  float *O1=(float*)(ws+O1o), *O2=(float*)(ws+O2o);

  // per-call zeroing of accumulators / padded tails
  {
    size_t n = ((size_t)BATCH*208*64*2)/4;
    k_zero<<<(unsigned)((n+255)/256),256,0,stream>>>((uint32_t*)Y4R, n);
    size_t m = (size_t)BATCH*368;
    k_zero<<<(unsigned)((m+255)/256),256,0,stream>>>((uint32_t*)C1, m);
    k_zero<<<(unsigned)((m+255)/256),256,0,stream>>>((uint32_t*)C2, m);
  }

  auto gemm = [&](const half_t* A, const half_t* B, void* D, const float* bias,
                  int M, int K, int N, int taps, int dil,
                  long long sA, long long sB, long long sD,
                  int act, int of32, int acc){
    int useLds = ((long long)taps*M*K <= 32768 && sA == 0) ? 1 : 0;
    dim3 grd((unsigned)((N+63)/64), (unsigned)((M+63)/64), BATCH);
    k_gemm<<<grd,256,0,stream>>>(A,B,D,bias,M,K,N,taps,dil,sA,sB,sD,act,of32,acc,useLds);
  };
  const long long RS64 = (long long)ROWS*64, RS128 = (long long)ROWS*128;
  const int pools[4] = {4,5,2,2};
  const int dils[7]  = {1,2,4,8,16,32,64};

  auto encode = [&](const float* x, const int* fb, float* hf, float* hb){
    k_sinc<<<(unsigned)(((size_t)BATCH*T0*64+255)/256),256,0,stream>>>(x, PF(SINC_W), PF(SINC_B), SIN);
    int T = T0;
    for (int s=0;s<4;++s){
      gemm(WH+st_first[s], SIN, XB, PF(WN(s,1)), 64,64,T,1,1, 0,RS64,RS64, 0,0,0);
      k_zero_rows<<<(BATCH*128*64)/256,256,0,stream>>>(XB, (size_t)ROWS*64, (size_t)T, 128*64);
      for (int k=0;k<7;++k){
        size_t gsz = (size_t)BATCH*T*64;
        gemm(WH+st_in[s][k], XB, G, PF(WN(s,2+6*k+1)), 128,64,T,3,dils[k], 0,RS64,RS128, 0,0,0);
        k_gate<<<(unsigned)((gsz+255)/256),256,0,stream>>>(G, U, T);
        gemm(WH+st_skip[s][k], U, SK, PF(WN(s,2+6*k+3)), 64,64,T,1,1, 0,RS64,RS64, 0,0, k>0?1:0);
        gemm(WH+st_res[s][k],  U, RT, PF(WN(s,2+6*k+5)), 64,64,T,1,1, 0,RS64,RS64, 0,0,0);
        k_resid<<<(unsigned)((gsz+255)/256),256,0,stream>>>(XB, RT, T);
      }
      {
        size_t gsz = (size_t)BATCH*T*64;
        k_relu16<<<(unsigned)((gsz+255)/256),256,0,stream>>>(SK, T);
      }
      gemm(WH+st_lc1[s], SK, U, PF(WN(s,45)), 64,64,T,1,1, 0,RS64,RS64, 1,0,0);
      gemm(WH+st_lc2[s], U, RT, PF(WN(s,47)), 64,64,T,1,1, 0,RS64,RS64, 0,0,0);
      int Tn = T / pools[s];
      size_t psz = (size_t)BATCH*Tn*64;
      k_maxpool<<<(unsigned)((psz+255)/256),256,0,stream>>>(RT, SIN, Tn, pools[s]);
      T = Tn;
    }
    // BiLSTM (T == 200 here; GEMM N padded to 208)
    gemm(WH+w_ih_f, SIN, XW, (const float*)(ws+BSF), 512,64,208,1,1,
         0, RS64, 208*512, 0,1,0);
    k_lstm<<<BATCH,128,0,stream>>>(XW, WH+w_hh_f, hf, 200);
    k_rev<<<(BATCH*200*64)/256,256,0,stream>>>(SIN, Y4R, fb);
    gemm(WH+w_ih_b, Y4R, XW, (const float*)(ws+BSB), 512,64,208,1,1,
         0, 208*64, 208*512, 0,1,0);
    k_lstm<<<BATCH,128,0,stream>>>(XW, WH+w_hh_b, HBR, 200);
    k_unrev<<<(BATCH*200*128)/256,256,0,stream>>>(HBR, hb, fb);
  };

  encode(x1, (const int*)(ws+F1), HF1, HB1);
  encode(x2, (const int*)(ws+F2), HF2, HB2);

  auto lmemb = [&](const float* wf, float* wemb){
    size_t n = (size_t)BATCH*160*768;
    k_cvt<<<(unsigned)((n+255)/256),256,0,stream>>>(wf, WLH, n);
    gemm(WH+w_lm1, WLH, LMH, PF(LM0+1), 256,768,160,1,1, 0, 160*768, 160*256, 1,0,0);
    gemm(WH+w_lm2, LMH, wemb, PF(LM0+3), 256,256,160,1,1, 0, 160*256, 160*256, 0,1,0);
  };
  lmemb(wavlm1, WEMB1);
  lmemb(wavlm2, WEMB2);

  k_build_n<<<(unsigned)(((size_t)BATCH*368*256)/256),256,0,stream>>>(
      HF1, HB1, WEMB1, (const int*)(ws+F1), PI(LENLM1), N1, N1H);
  k_build_n<<<(unsigned)(((size_t)BATCH*368*256)/256),256,0,stream>>>(
      HF2, HB2, WEMB2, (const int*)(ws+F2), PI(LENLM2), N2, N2H);

  // scores[b][i][j] = n2[i] . n1[j]
  gemm(N1H, N2H, SC, nullptr, 368,256,368,1,1,
       (long long)368*256, (long long)368*256, (long long)368*368, 0,1,0);

  const int* L1i = (const int*)(ws+L1);
  const int* L2i = (const int*)(ws+L2);
  k_rowsm<<<dim3(368,BATCH),256,0,stream>>>(SC, L1i, L2i, C1);
  k_colsm<<<dim3(368,BATCH),256,0,stream>>>(SC, L1i, L2i, C2);
  k_dvec<<<BATCH,256,0,stream>>>(C1, N2, N1, L1i, D1);
  k_dvec<<<BATCH,256,0,stream>>>(C2, N1, N2, L2i, D2);

  k_head<<<BATCH,128,0,stream>>>(D1, PF(DV0+0),PF(DV0+1),PF(DV0+2),PF(DV0+3), O1);
  k_head<<<BATCH,128,0,stream>>>(D2, PF(DV0+0),PF(DV0+1),PF(DV0+2),PF(DV0+3), O2);
  k_final<<<1,BATCH,0,stream>>>(O1, O2, (float*)d_out);
}
